// CrossAttention_8169027797805
// MI455X (gfx1250) — matrix operand order repaired and checked by reading, not compiled, not run on hardware
//
#include <hip/hip_runtime.h>
#include <hip/hip_bf16.h>
#include <cstdint>

// ---------------------------------------------------------------------------
// CrossAttention for MI455X (gfx1250, wave32, WMMA).
// Pipeline (all on `stream`):
//   1) f32->f16 convert of inputs/context; f32->f16 transpose of Wq/Wk/Wv/Wo
//   2) Q = x @ Wq + bq, K = c @ Wk + bk      (f16 WMMA GEMM, f16 out)
//   3) V = c @ Wv + bv, stored transposed per head [B,H,HS,SK] (f16)
//   4) flash attention per (b,h,16-q-rows): online softmax, never
//      materializes the 536MB score tensor
//   5) out = attn @ Wo + bo (f32 out, written straight to d_out)
// Sizes fixed by the reference: B=2, SQ=SK=2048, D=1024, H=16, HS=64.
// GEMM tile staging uses GLOBAL_LOAD_ASYNC_TO_LDS_B128 via inline asm
// (ASYNCcnt path); the clang builtin's pointer params are typed in the CUDA
// __device__/__shared__ address spaces which HIP source cannot spell.
// ---------------------------------------------------------------------------

typedef _Float16 f16;
typedef __attribute__((ext_vector_type(16))) _Float16 v16h;
typedef __attribute__((ext_vector_type(8)))  _Float16 v8h;
typedef __attribute__((ext_vector_type(4)))  _Float16 v4h;
typedef __attribute__((ext_vector_type(8)))  float    v8f;

#define CA_B   2
#define CA_SQ  2048
#define CA_SK  2048
#define CA_D   1024
#define CA_H   16
#define CA_HS  64
#define CA_M   (CA_B * CA_SQ)      // 4096 rows for every projection GEMM

// Async DMA of 16 bytes global -> LDS (per lane). VDST = LDS byte offset
// (low 32 bits of a generic LDS pointer), VADDR = 64-bit global address.
__device__ __forceinline__ void async_ld_b128(const f16* gptr, f16* lptr) {
  const uint32_t laddr = (uint32_t)(uintptr_t)lptr;
  const uint64_t gaddr = (uint64_t)(uintptr_t)gptr;
  asm volatile("global_load_async_to_lds_b128 %0, %1, off"
               :: "v"(laddr), "v"(gaddr) : "memory");
}

__device__ __forceinline__ void wait_async0() {
#if __has_builtin(__builtin_amdgcn_s_wait_asynccnt)
  __builtin_amdgcn_s_wait_asynccnt(0);
#else
  asm volatile("s_wait_asynccnt 0" ::: "memory");
#endif
}

// Build a 16-half WMMA operand from two 16-byte chunks (global or LDS).
__device__ __forceinline__ v16h ld16(const f16* p0, const f16* p1) {
  v16h r;
  *reinterpret_cast<v8h*>(&r)       = *reinterpret_cast<const v8h*>(p0);
  *(reinterpret_cast<v8h*>(&r) + 1) = *reinterpret_cast<const v8h*>(p1);
  return r;
}

__device__ __forceinline__ v8f wmma_f16(v16h a, v16h b, v8f c) {
  return __builtin_amdgcn_wmma_f32_16x16x32_f16(false, a, false, b,
                                                (short)0, c, false, false);
}

// ---------------------------------------------------------------------------
// f32 -> f16 elementwise convert (4 elements / thread, exact sizes)
// ---------------------------------------------------------------------------
__global__ __launch_bounds__(256) void cvt_f32_f16(const float* __restrict__ in,
                                                   f16* __restrict__ out) {
  int i = blockIdx.x * blockDim.x + threadIdx.x;
  float4 v = reinterpret_cast<const float4*>(in)[i];
  v4h h;
  h.x = (f16)v.x; h.y = (f16)v.y; h.z = (f16)v.z; h.w = (f16)v.w;
  reinterpret_cast<v4h*>(out)[i] = h;
}

// ---------------------------------------------------------------------------
// W[1024][1024] f32 (in,out) -> Wt[1024][1024] f16 (out,in), LDS-tiled
// ---------------------------------------------------------------------------
__global__ __launch_bounds__(256) void transpose_w(const float* __restrict__ W,
                                                   f16* __restrict__ Wt) {
  __shared__ float tile[32][33];
  const int tx = threadIdx.x, ty = threadIdx.y;
  const int xs = blockIdx.x * 32 + tx;
#pragma unroll
  for (int i = ty; i < 32; i += 8)
    tile[i][tx] = W[(size_t)(blockIdx.y * 32 + i) * CA_D + xs];
  __syncthreads();
  const int xo = blockIdx.y * 32 + tx;
#pragma unroll
  for (int i = ty; i < 32; i += 8)
    Wt[(size_t)(blockIdx.x * 32 + i) * CA_D + xo] = (f16)tile[tx][i];
}

// ---------------------------------------------------------------------------
// GEMM: C[4096,1024] = A[4096,1024] @ Bt[1024,1024]^T + bias
// 128x128 macro-tile, 8 waves of 64x32, K-step 32, f16 WMMA, f32 accum.
// MODE 0: f16 row-major out (Q, K, attn-input)
// MODE 1: f16 out scattered to per-head transposed V layout [B,H,HS,SK]
// MODE 2: f32 row-major out (final projection -> d_out)
// ---------------------------------------------------------------------------
template <int MODE>
__global__ __launch_bounds__(256) void gemm_wmma(const f16* __restrict__ A,
                                                 const f16* __restrict__ Bt,
                                                 const float* __restrict__ bias,
                                                 void* __restrict__ outp) {
  __shared__ f16 As[128][40];   // 40-half row stride: 16B-aligned, bank-skewed
  __shared__ f16 Bs[128][40];
  const int m0 = blockIdx.x * 128;
  const int n0 = blockIdx.y * 128;
  const int t  = threadIdx.x;
  const int w  = t >> 5, l = t & 31, lr = l & 15, hi = l >> 4;
  const int wm = (w & 1) * 64;        // 2 waves along M
  const int wn = (w >> 1) * 32;       // 4 waves along N

  const v8f z8 = {0.f, 0.f, 0.f, 0.f, 0.f, 0.f, 0.f, 0.f};
  v8f acc[4][2];
#pragma unroll
  for (int mi = 0; mi < 4; ++mi)
#pragma unroll
    for (int ni = 0; ni < 2; ++ni) acc[mi][ni] = z8;

  for (int kb = 0; kb < CA_D; kb += 32) {
#pragma unroll
    for (int i = 0; i < 2; ++i) {           // 512 v8h chunks / 256 threads
      const int lin = t + i * 256;
      const int r = lin >> 2;
      const int c = (lin & 3) * 8;
      // ASYNCcnt DMA straight into LDS, no VGPR staging.
      async_ld_b128(A  + (size_t)(m0 + r) * CA_D + kb + c, &As[r][c]);
      async_ld_b128(Bt + (size_t)(n0 + r) * CA_D + kb + c, &Bs[r][c]);
    }
    wait_async0();
    __syncthreads();

    v16h af[4], bf[2];
#pragma unroll
    for (int mi = 0; mi < 4; ++mi) {
      const f16* p = &As[wm + mi * 16 + lr][hi * 8];
      af[mi] = ld16(p, p + 16);
    }
#pragma unroll
    for (int ni = 0; ni < 2; ++ni) {
      const f16* p = &Bs[wn + ni * 16 + lr][hi * 8];
      bf[ni] = ld16(p, p + 16);
    }
#pragma unroll
    for (int mi = 0; mi < 4; ++mi)
#pragma unroll
      for (int ni = 0; ni < 2; ++ni)
        acc[mi][ni] = wmma_f16(af[mi], bf[ni], acc[mi][ni]);
    __syncthreads();
  }

  // Epilogue. C/D layout: VGPR j holds (M = j + 8*hi, N = lr) of each tile.
#pragma unroll
  for (int mi = 0; mi < 4; ++mi) {
#pragma unroll
    for (int ni = 0; ni < 2; ++ni) {
      const int n = n0 + wn + ni * 16 + lr;
      const float bv = bias[n];
#pragma unroll
      for (int j = 0; j < 8; ++j) {
        const int m = m0 + wm + mi * 16 + j + hi * 8;
        const float val = acc[mi][ni][j] + bv;
        if (MODE == 0) {
          reinterpret_cast<f16*>(outp)[(size_t)m * CA_D + n] = (f16)val;
        } else if (MODE == 1) {
          const int bb = m >> 11, s = m & (CA_SK - 1);
          const int hh = n >> 6,  d = n & (CA_HS - 1);
          reinterpret_cast<f16*>(outp)
              [(((size_t)(bb * CA_H + hh) * CA_HS + d) << 11) + s] = (f16)val;
        } else {
          reinterpret_cast<float*>(outp)[(size_t)m * CA_D + n] = val;
        }
      }
    }
  }
}

// ---------------------------------------------------------------------------
// Flash attention. Block = 128 threads = 4 waves; each wave owns 16 Q rows of
// one (b,h) and streams all SK keys in chunks of 32. Online softmax keeps the
// per-row (max, sum) replicated across each 16-lane half-group (matching the
// C/D layout row ownership). P is restaged A-layout through per-wave LDS.
// ---------------------------------------------------------------------------
__global__ __launch_bounds__(128) void attn_flash(const f16* __restrict__ Q,
                                                  const f16* __restrict__ Kc,
                                                  const f16* __restrict__ Vt,
                                                  f16* __restrict__ Oa) {
  __shared__ f16 Pt[4][2][16][24];   // [wave][key-subtile][row][col(+pad)]
  const int t  = threadIdx.x, w = t >> 5, l = t & 31, lr = l & 15, hi = l >> 4;
  const int b  = blockIdx.z, h = blockIdx.y;
  const int q0 = blockIdx.x * 64 + w * 16;
  const float C = 0.125f * 1.44269504088896f;  // HS^-0.5 * log2(e)

  // Q fragments (16 rows x 64 head-dims = two k=32 A-operands), loaded once.
  const f16* qp = Q + (size_t)(b * CA_SQ + q0 + lr) * CA_D + h * CA_HS;
  v16h aq[2];
  aq[0] = ld16(qp + hi * 8,      qp + hi * 8 + 16);
  aq[1] = ld16(qp + 32 + hi * 8, qp + 32 + hi * 8 + 16);

  const f16* kp0 = Kc + (size_t)b * CA_SK * CA_D + h * CA_HS;   // [key][dim]
  const f16* vp0 = Vt + (size_t)(b * CA_H + h) * CA_HS * CA_SK; // [dim][key]

  const v8f z8 = {0.f, 0.f, 0.f, 0.f, 0.f, 0.f, 0.f, 0.f};
  v8f o[4] = {z8, z8, z8, z8};
  float mo[8], lo[8];
#pragma unroll
  for (int j = 0; j < 8; ++j) { mo[j] = -1e30f; lo[j] = 0.f; }

  for (int kb = 0; kb < CA_SK; kb += 32) {
    // ---- S = Q K^T for 32 keys (two 16x16 tiles, k=64 in two WMMA steps)
    v16h kf[2][2];
#pragma unroll
    for (int t2 = 0; t2 < 2; ++t2) {
      const f16* kp = kp0 + (size_t)(kb + t2 * 16 + lr) * CA_D;
      kf[t2][0] = ld16(kp + hi * 8,      kp + hi * 8 + 16);
      kf[t2][1] = ld16(kp + 32 + hi * 8, kp + 32 + hi * 8 + 16);
    }
    v8f s0 = z8, s1 = z8;
    s0 = wmma_f16(aq[0], kf[0][0], s0);
    s0 = wmma_f16(aq[1], kf[0][1], s0);
    s1 = wmma_f16(aq[0], kf[1][0], s1);
    s1 = wmma_f16(aq[1], kf[1][1], s1);

    if (kb + 32 < CA_SK) {  // hint next chunk into cache (global_prefetch_b8)
      __builtin_prefetch(kp0 + (size_t)(kb + 32 + lr) * CA_D, 0, 1);
      __builtin_prefetch(vp0 + (size_t)lr * CA_SK + kb + 32, 0, 1);
    }

    // ---- online softmax over these 32 keys (v_exp_f32 via exp2 builtin)
    float tj[8], p0a[8], p1a[8], corr[8];
#pragma unroll
    for (int j = 0; j < 8; ++j) tj[j] = fmaxf(s0[j], s1[j]);
#pragma unroll
    for (int off = 1; off < 16; off <<= 1)   // row max within 16-lane half
#pragma unroll
      for (int j = 0; j < 8; ++j) tj[j] = fmaxf(tj[j], __shfl_xor(tj[j], off, 32));
#pragma unroll
    for (int j = 0; j < 8; ++j) {
      const float mn = fmaxf(mo[j], tj[j]);
      corr[j] = __builtin_amdgcn_exp2f((mo[j] - mn) * C);
      p0a[j]  = __builtin_amdgcn_exp2f((s0[j] - mn) * C);
      p1a[j]  = __builtin_amdgcn_exp2f((s1[j] - mn) * C);
      mo[j]   = mn;
      tj[j]   = p0a[j] + p1a[j];
    }
#pragma unroll
    for (int off = 1; off < 16; off <<= 1)   // row sum within 16-lane half
#pragma unroll
      for (int j = 0; j < 8; ++j) tj[j] += __shfl_xor(tj[j], off, 32);
#pragma unroll
    for (int j = 0; j < 8; ++j) lo[j] = lo[j] * corr[j] + tj[j];
#pragma unroll
    for (int ot = 0; ot < 4; ++ot)
#pragma unroll
      for (int j = 0; j < 8; ++j) o[ot][j] *= corr[j];

    // ---- restage P (C/D layout) -> A layout via per-wave LDS tile
#pragma unroll
    for (int j = 0; j < 8; ++j) {
      Pt[w][0][j + hi * 8][lr] = (f16)p0a[j];
      Pt[w][1][j + hi * 8][lr] = (f16)p1a[j];
    }
    const v16h ap = ld16(&Pt[w][0][lr][hi * 8], &Pt[w][1][lr][hi * 8]);

    // ---- O += P V   (B-operand contiguous from transposed V [dim][key])
#pragma unroll
    for (int ot = 0; ot < 4; ++ot) {
      const f16* vp = vp0 + (size_t)(ot * 16 + lr) * CA_SK + kb + hi * 8;
      const v16h bv = ld16(vp, vp + 16);
      o[ot] = wmma_f16(ap, bv, o[ot]);
    }
  }

  // ---- normalize (v_rcp_f32 + mul, no IEEE divide chain) and store f16
  float inv[8];
#pragma unroll
  for (int j = 0; j < 8; ++j) inv[j] = __builtin_amdgcn_rcpf(lo[j]);
#pragma unroll
  for (int ot = 0; ot < 4; ++ot)
#pragma unroll
    for (int j = 0; j < 8; ++j) {
      const int q = q0 + j + hi * 8;
      const int d = ot * 16 + lr;
      Oa[(size_t)(b * CA_SQ + q) * CA_D + h * CA_HS + d] =
          (f16)(o[ot][j] * inv[j]);
    }
}

// ---------------------------------------------------------------------------
// Host-side launcher
// ---------------------------------------------------------------------------
extern "C" void kernel_launch(void* const* d_in, const int* in_sizes, int n_in,
                              void* d_out, int out_size, void* d_ws, size_t ws_size,
                              hipStream_t stream) {
  (void)in_sizes; (void)n_in; (void)out_size; (void)ws_size;
  const float* inputs  = (const float*)d_in[0];
  const float* context = (const float*)d_in[1];
  const float* Wq = (const float*)d_in[2]; const float* bq = (const float*)d_in[3];
  const float* Wk = (const float*)d_in[4]; const float* bk = (const float*)d_in[5];
  const float* Wv = (const float*)d_in[6]; const float* bv = (const float*)d_in[7];
  const float* Wo = (const float*)d_in[8]; const float* bo = (const float*)d_in[9];

  char* ws = (char*)d_ws;
  const size_t SZ_ACT = (size_t)CA_M * CA_D * sizeof(f16);   // 8 MiB
  const size_t SZ_W   = (size_t)CA_D * CA_D * sizeof(f16);   // 2 MiB
  f16* x16  = (f16*)(ws);
  f16* c16  = (f16*)(ws + SZ_ACT);
  f16* wqt  = (f16*)(ws + 2 * SZ_ACT);
  f16* wkt  = (f16*)(ws + 2 * SZ_ACT + SZ_W);
  f16* wvt  = (f16*)(ws + 2 * SZ_ACT + 2 * SZ_W);
  f16* wot  = (f16*)(ws + 2 * SZ_ACT + 3 * SZ_W);
  f16* q16  = (f16*)(ws + 2 * SZ_ACT + 4 * SZ_W);
  f16* k16  = (f16*)(ws + 3 * SZ_ACT + 4 * SZ_W);
  f16* vt16 = (f16*)(ws + 4 * SZ_ACT + 4 * SZ_W);
  f16* a16  = (f16*)(ws + 5 * SZ_ACT + 4 * SZ_W);

  // 1) convert activations + transpose weights to f16
  const int nact4 = (CA_M * CA_D) / 4;                       // 1,048,576
  cvt_f32_f16<<<nact4 / 256, 256, 0, stream>>>(inputs,  x16);
  cvt_f32_f16<<<nact4 / 256, 256, 0, stream>>>(context, c16);
  dim3 tb(32, 8), tg(32, 32);
  transpose_w<<<tg, tb, 0, stream>>>(Wq, wqt);
  transpose_w<<<tg, tb, 0, stream>>>(Wk, wkt);
  transpose_w<<<tg, tb, 0, stream>>>(Wv, wvt);
  transpose_w<<<tg, tb, 0, stream>>>(Wo, wot);

  // 2-3) projections
  dim3 gg(CA_M / 128, CA_D / 128);                           // 32 x 8
  gemm_wmma<0><<<gg, 256, 0, stream>>>(x16, wqt, bq, (void*)q16);
  gemm_wmma<0><<<gg, 256, 0, stream>>>(c16, wkt, bk, (void*)k16);
  gemm_wmma<1><<<gg, 256, 0, stream>>>(c16, wvt, bv, (void*)vt16);

  // 4) flash attention
  dim3 ag(CA_SQ / 64, CA_H, CA_B);                           // 32 x 16 x 2
  attn_flash<<<ag, 128, 0, stream>>>(q16, k16, vt16, a16);

  // 5) output projection straight into d_out (f32)
  gemm_wmma<2><<<gg, 256, 0, stream>>>(a16, wot, bo, d_out);
}
